// ItemEmbeddingLayer_26517128085577
// MI455X (gfx1250) — compile-verified, matrix operand order checked by reading, not run on hardware
//
#include <hip/hip_runtime.h>

#define NI    100000
#define NOPS  200000
#define EDGES 600000
#define DIM   128
#define HID   256
#define HID2  128
#define OUTD  128

typedef __attribute__((ext_vector_type(16))) __bf16 v16bf;
typedef __attribute__((ext_vector_type(8)))  float  v8f;

union BFrag { v16bf v; unsigned u[8]; unsigned short us[16]; };
union AccF  { v8f  v; float f[8]; };

// round-to-nearest-even f32 -> bf16
__device__ __forceinline__ unsigned short f2bf(float x) {
  union { float f; unsigned u; } c; c.f = x;
  unsigned r = c.u + 0x7FFFu + ((c.u >> 16) & 1u);
  return (unsigned short)(r >> 16);
}
__device__ __forceinline__ float elu_f(float x) {
  return x > 0.f ? x : (__expf(x) - 1.f);
}
// K offset of packed pair v (0..7) for lane-half h, within a 16x32 16-bit fragment
__device__ __forceinline__ int kb(int v, int h) {
  return ((v & 4) ? 16 : 0) + 2 * (v & 3) + 8 * h;
}

// ---------------- zero fill ----------------
__global__ void zero_f4(float4* p, long n4) {
  long i = (long)blockIdx.x * blockDim.x + threadIdx.x;
  if (i < n4) p[i] = make_float4(0.f, 0.f, 0.f, 0.f);
}

// ------- pack W[K x N] f32 row-major -> bf16 fragment-ordered tiles -------
// tile(nt,ks) = 512 contiguous bf16: lane-major (lane*16 + e)
__global__ void pack_w(const float* __restrict__ W, unsigned short* __restrict__ out,
                       int K, int N) {
  int t = blockIdx.x * blockDim.x + threadIdx.x;
  if (t >= K * N) return;
  int e    = t & 15;
  int lane = (t >> 4) & 31;
  int tile = t >> 9;
  int NK = K / 32;
  int nt = tile / NK, ks = tile % NK;
  int half = lane >> 4, colp = lane & 15;
  int k = ks * 32 + kb(e >> 1, half) + (e & 1);
  int n = nt * 16 + colp;
  out[t] = f2bf(W[(size_t)k * N + n]);
}

// ------- segment-sum scatter: 32 lanes per edge, 4 floats each -------
__global__ void scatter_add(const float* __restrict__ feat, const int* __restrict__ eidx,
                            float* __restrict__ agg, int nedges) {
  int gid = blockIdx.x * blockDim.x + threadIdx.x;
  int e = gid >> 5, c = gid & 31;
  if (e >= nedges) return;
  int seg = eidx[e];            // edge_index[0][e] : segment id
  int src = eidx[nedges + e];   // edge_index[1][e] : gathered row
  const float4 v = *(const float4*)(feat + (size_t)src * DIM + c * 4);
  float* dst = agg + (size_t)seg * DIM + c * 4;
  unsafeAtomicAdd(dst + 0, v.x);
  unsafeAtomicAdd(dst + 1, v.y);
  unsafeAtomicAdd(dst + 2, v.z);
  unsafeAtomicAdd(dst + 3, v.w);
}

// ------- fragment loaders -------
__device__ __forceinline__ void load_wfrag(BFrag& b, const unsigned short* __restrict__ Wp,
                                           int tile, int lane) {
  const uint4* wp = ((const uint4*)(Wp + (size_t)tile * 512)) + lane * 2;
  uint4 w0 = wp[0], w1 = wp[1];
  b.u[0] = w0.x; b.u[1] = w0.y; b.u[2] = w0.z; b.u[3] = w0.w;
  b.u[4] = w1.x; b.u[5] = w1.y; b.u[6] = w1.z; b.u[7] = w1.w;
}
__device__ __forceinline__ void load_afrag_lds(BFrag& a, const unsigned short* h,
                                               int K, int col, int half, int ks) {
  const uint4* hp = (const uint4*)h;
  uint4 c0 = hp[(col * K + ks * 32 + half * 8) >> 3];
  uint4 c1 = hp[(col * K + ks * 32 + 16 + half * 8) >> 3];
  a.u[0] = c0.x; a.u[1] = c0.y; a.u[2] = c0.z; a.u[3] = c0.w;
  a.u[4] = c1.x; a.u[5] = c1.y; a.u[6] = c1.z; a.u[7] = c1.w;
}
__device__ __forceinline__ void cvt8(unsigned short* dst, float4 a, float4 b) {
  dst[0] = f2bf(a.x); dst[1] = f2bf(a.y); dst[2] = f2bf(a.z); dst[3] = f2bf(a.w);
  dst[4] = f2bf(b.x); dst[5] = f2bf(b.y); dst[6] = f2bf(b.z); dst[7] = f2bf(b.w);
}

// ------- 3-layer MLP: IN_DIM -> 256 (ELU) -> 128 (ELU) -> 128 -------
// wave handles 16 rows; 4 waves/WG; per-wave LDS slice for hidden transpose
template <int IN_DIM>
__global__ __launch_bounds__(128) void mlp3_wmma(
    const float* __restrict__ X, const int* __restrict__ rowidx, int n_rows,
    const unsigned short* __restrict__ W1p, const float* __restrict__ b1,
    const unsigned short* __restrict__ W2p, const float* __restrict__ b2,
    const unsigned short* __restrict__ W3p, const float* __restrict__ b3,
    float* __restrict__ out, int out_ld, int out_coloff, int store_rows) {
  constexpr int NK1 = IN_DIM / 32;
  constexpr int WAVES = 4;
  __shared__ unsigned short hbuf[WAVES][16 * HID];  // 32 KB static

  const int wid  = threadIdx.x >> 5;
  const int lane = threadIdx.x & 31;
  const int half = lane >> 4;
  const int col  = lane & 15;
  const int m0   = (blockIdx.x * WAVES + wid) * 16;
  if (m0 >= n_rows) return;
  unsigned short* h1 = hbuf[wid];

  // ---- layer-1 A fragments: per-lane two contiguous 32B chunks per K-step ----
  BFrag a1[NK1];
  {
    int row = m0 + col;  // lanes 0-15 and 16-31 both map M = lane&15
    int g = rowidx ? rowidx[row] : row;
    const float* xr = X + (size_t)g * IN_DIM;
#pragma unroll
    for (int ks = 0; ks < NK1; ++ks) {
      const float4* c0 = (const float4*)(xr + ks * 32 + half * 8);
      const float4* c1 = (const float4*)(xr + ks * 32 + 16 + half * 8);
      cvt8(&a1[ks].us[0], c0[0], c0[1]);
      cvt8(&a1[ks].us[8], c1[0], c1[1]);
    }
  }

  // ---- layer 1: [16 x IN_DIM] @ [IN_DIM x 256] ----
#pragma unroll
  for (int nt = 0; nt < HID / 16; ++nt) {
    AccF acc;
#pragma unroll
    for (int r = 0; r < 8; ++r) acc.f[r] = 0.f;
#pragma unroll
    for (int ks = 0; ks < NK1; ++ks) {
      BFrag b; load_wfrag(b, W1p, nt * NK1 + ks, lane);
      acc.v = __builtin_amdgcn_wmma_f32_16x16x32_bf16(
          false, a1[ks].v, false, b.v, (short)0, acc.v, false, false);
    }
    float bv = b1[nt * 16 + col];
#pragma unroll
    for (int r = 0; r < 8; ++r) {
      int m = r + half * 8;  // C layout: vgpr r -> M=r (lanes 0-15) / M=r+8
      h1[m * HID + nt * 16 + col] = f2bf(elu_f(acc.f[r] + bv));
    }
  }
  __asm__ volatile("s_wait_dscnt 0" ::: "memory");

  // ---- layer 2: preload all A2 frags, then reuse LDS slice for h2 ----
  BFrag a2[HID / 32];
#pragma unroll
  for (int ks = 0; ks < HID / 32; ++ks) load_afrag_lds(a2[ks], h1, HID, col, half, ks);
  __asm__ volatile("s_wait_dscnt 0" ::: "memory");

#pragma unroll
  for (int nt = 0; nt < HID2 / 16; ++nt) {
    AccF acc;
#pragma unroll
    for (int r = 0; r < 8; ++r) acc.f[r] = 0.f;
#pragma unroll
    for (int ks = 0; ks < HID / 32; ++ks) {
      BFrag b; load_wfrag(b, W2p, nt * (HID / 32) + ks, lane);
      acc.v = __builtin_amdgcn_wmma_f32_16x16x32_bf16(
          false, a2[ks].v, false, b.v, (short)0, acc.v, false, false);
    }
    float bv = b2[nt * 16 + col];
#pragma unroll
    for (int r = 0; r < 8; ++r) {
      int m = r + half * 8;
      h1[m * HID2 + nt * 16 + col] = f2bf(elu_f(acc.f[r] + bv));  // h2 overlays h1
    }
  }
  __asm__ volatile("s_wait_dscnt 0" ::: "memory");

  BFrag a3[HID2 / 32];
#pragma unroll
  for (int ks = 0; ks < HID2 / 32; ++ks) load_afrag_lds(a3[ks], h1, HID2, col, half, ks);
  __asm__ volatile("s_wait_dscnt 0" ::: "memory");

  // ---- layer 3: store f32 to global at column offset ----
#pragma unroll
  for (int nt = 0; nt < OUTD / 16; ++nt) {
    AccF acc;
#pragma unroll
    for (int r = 0; r < 8; ++r) acc.f[r] = 0.f;
#pragma unroll
    for (int ks = 0; ks < HID2 / 32; ++ks) {
      BFrag b; load_wfrag(b, W3p, nt * (HID2 / 32) + ks, lane);
      acc.v = __builtin_amdgcn_wmma_f32_16x16x32_bf16(
          false, a3[ks].v, false, b.v, (short)0, acc.v, false, false);
    }
    float bv = b3[nt * 16 + col];
    int n = out_coloff + nt * 16 + col;
#pragma unroll
    for (int r = 0; r < 8; ++r) {
      int m = m0 + r + half * 8;
      if (m < store_rows) out[(size_t)m * out_ld + n] = acc.f[r] + bv;
    }
  }
}

extern "C" void kernel_launch(void* const* d_in, const int* in_sizes, int n_in,
                              void* d_out, int out_size, void* d_ws, size_t ws_size,
                              hipStream_t stream) {
  const float* items      = (const float*)d_in[0];
  const int*   parents    = (const int*)d_in[1];
  const float* operations = (const float*)d_in[2];
  const int*   item_ei    = (const int*)d_in[3];
  const int*   op_ei      = (const int*)d_in[4];
  const float* P[5][6];
  for (int i = 0; i < 5; ++i)
    for (int j = 0; j < 6; ++j) P[i][j] = (const float*)d_in[5 + i * 6 + j];
  float* out = (float*)d_out;

  // workspace carve-out
  char* ws = (char*)d_ws;
  size_t cur = 0;
  auto alloc = [&](size_t bytes) -> char* {
    char* p = ws + cur;
    cur += (bytes + 255) & ~(size_t)255;
    return p;
  };
  float* aggC   = (float*)alloc((size_t)NI * DIM * 4);
  float* aggO   = (float*)alloc((size_t)NI * DIM * 4);
  float* concat = (float*)alloc((size_t)NI * 4 * OUTD * 4);
  const int indim[5] = {DIM, DIM, DIM, DIM, 4 * OUTD};
  unsigned short* Wp[5][3];
  for (int i = 0; i < 5; ++i) {
    Wp[i][0] = (unsigned short*)alloc((size_t)indim[i] * HID * 2);
    Wp[i][1] = (unsigned short*)alloc((size_t)HID * HID2 * 2);
    Wp[i][2] = (unsigned short*)alloc((size_t)HID2 * OUTD * 2);
  }

  // 1) zero aggregation buffers (contiguous) + last output row
  {
    long n4 = (long)2 * NI * DIM / 4;
    zero_f4<<<(unsigned)((n4 + 255) / 256), 256, 0, stream>>>((float4*)aggC, n4);
    zero_f4<<<1, 64, 0, stream>>>((float4*)(out + (size_t)(NI - 1) * OUTD), OUTD / 4);
  }

  // 2) pack weights into bf16 fragment layout
  for (int i = 0; i < 5; ++i) {
    const int K[3] = {indim[i], HID, HID2};
    const int N[3] = {HID, HID2, OUTD};
    for (int j = 0; j < 3; ++j) {
      int total = K[j] * N[j];
      pack_w<<<(total + 255) / 256, 256, 0, stream>>>(P[i][2 * j], Wp[i][j], K[j], N[j]);
    }
  }

  // 3) segment sums via fp32 global atomics
  {
    int blocks = (EDGES * 32) / 256;
    scatter_add<<<blocks, 256, 0, stream>>>(items, item_ei, aggC, EDGES);
    scatter_add<<<blocks, 256, 0, stream>>>(operations, op_ei, aggO, EDGES);
  }

  // 4) four 128-in MLPs -> concat[N,512] at column offsets (parent|child|ops|self)
  {
    int ntiles = (NI + 15) / 16;           // 6250
    int blocks = (ntiles + 3) / 4;         // 4 waves/WG
    mlp3_wmma<DIM><<<blocks, 128, 0, stream>>>(
        items, parents, NI, Wp[1][0], P[1][1], Wp[1][1], P[1][3], Wp[1][2], P[1][5],
        concat, 4 * OUTD, 0 * OUTD, NI);
    mlp3_wmma<DIM><<<blocks, 128, 0, stream>>>(
        aggC, nullptr, NI, Wp[2][0], P[2][1], Wp[2][1], P[2][3], Wp[2][2], P[2][5],
        concat, 4 * OUTD, 1 * OUTD, NI);
    mlp3_wmma<DIM><<<blocks, 128, 0, stream>>>(
        aggO, nullptr, NI, Wp[3][0], P[3][1], Wp[3][1], P[3][3], Wp[3][2], P[3][5],
        concat, 4 * OUTD, 2 * OUTD, NI);
    mlp3_wmma<DIM><<<blocks, 128, 0, stream>>>(
        items, nullptr, NI, Wp[0][0], P[0][1], Wp[0][1], P[0][3], Wp[0][2], P[0][5],
        concat, 4 * OUTD, 3 * OUTD, NI);
    // 5) combiner MLP (512-in); store only rows < N-1 (last row stays zero)
    mlp3_wmma<4 * OUTD><<<blocks, 128, 0, stream>>>(
        concat, nullptr, NI, Wp[4][0], P[4][1], Wp[4][1], P[4][3], Wp[4][2], P[4][5],
        out, OUTD, 0, NI - 1);
  }
}
